// RAPMemory_33474975105524
// MI455X (gfx1250) — matrix operand order repaired and checked by reading, not compiled, not run on hardware
//
#include <hip/hip_runtime.h>

// ---------------------------------------------------------------------------
// RAPMemory on MI455X (gfx1250):
//  - LTC scan: trans-pipe bound; 8 WGs/batch (128 WGs) with L2 ping-pong state
//    exchange + per-batch device barriers (1 per ODE unfold).
//  - GEMM tail: folded Hopfield (P/U), v_wmma_f32_16x16x32_bf16, TDM tile
//    stage-in via tensor_load_to_lds + s_wait_tensorcnt.
// ---------------------------------------------------------------------------

#define LOG2E 1.44269504088896f

typedef __bf16 v16bf __attribute__((ext_vector_type(16)));
typedef float  v8f   __attribute__((ext_vector_type(8)));

#if defined(__gfx1250__) && __has_builtin(__builtin_amdgcn_tensor_load_to_lds) && __has_builtin(__builtin_amdgcn_s_wait_tensorcnt)
#define USE_TDM 1
#else
#define USE_TDM 0
#endif

// sigmoid(arg) where t = -arg*log2(e) already:  1 / (1 + 2^t)
__device__ __forceinline__ float sigmoid_from_t(float t) {
  return __builtin_amdgcn_rcpf(1.0f + __builtin_amdgcn_exp2f(t));
}

__device__ __forceinline__ float softplus_f(float x) {
  return (x > 20.0f) ? x : log1pf(expf(x));
}

// ============================ setup kernels ================================

// Pack LTC params: {-sigma*log2e, sigma*mu*log2e, softplus(w)*mask, erev}
__global__ void prep_params(const float* __restrict__ smu, const float* __restrict__ ssig,
                            const float* __restrict__ sw,  const float* __restrict__ smask,
                            const float* __restrict__ serev,
                            const float* __restrict__ rmu, const float* __restrict__ rsig,
                            const float* __restrict__ rw,  const float* __restrict__ rmask,
                            const float* __restrict__ rerev,
                            const float* __restrict__ gleak, const float* __restrict__ cm,
                            float4* __restrict__ sens, float4* __restrict__ rec,
                            float* __restrict__ glsp, float* __restrict__ cmsp)
{
  int t = blockIdx.x * 256 + threadIdx.x;
  const int NS = 576 * 512, NR = 512 * 512;
  if (t < NS) {
    float sg = ssig[t], m = smu[t];
    float we = softplus_f(sw[t]) * smask[t];
    sens[t] = make_float4(-sg * LOG2E, sg * m * LOG2E, we, serev[t]);
  } else if (t < NS + NR) {
    int u = t - NS;
    float sg = rsig[u], m = rmu[u];
    float we = softplus_f(rw[u]) * rmask[u];
    rec[u] = make_float4(-sg * LOG2E, sg * m * LOG2E, we, rerev[u]);
  } else if (t < NS + NR + 512) {
    int u = t - NS - NR;
    glsp[u] = softplus_f(gleak[u]);
    cmsp[u] = softplus_f(cm[u]);
  }
}

// zero the device-barrier counters each launch (graph-replay safe)
__global__ void init_sync(int* __restrict__ cnt) {
  if (threadIdx.x < 16) cnt[threadIdx.x] = 0;
}

// k = lookup @ Wk, val = targets @ Wv   (each 32x256)
__global__ void prep_kv(const float* __restrict__ lookup, const float* __restrict__ targets,
                        const float* __restrict__ Wk, const float* __restrict__ Wv,
                        float* __restrict__ kmat, float* __restrict__ vmat)
{
  int t = blockIdx.x * 256 + threadIdx.x;       // 0..16383
  int which = t >> 13;
  int idx = t & 8191;
  int j = idx >> 8, c2 = idx & 255;
  const float* src = which ? targets : lookup;
  const float* W   = which ? Wv : Wk;
  float acc = 0.f;
  for (int c = 0; c < 256; ++c) acc = fmaf(src[j * 256 + c], W[c * 256 + c2], acc);
  (which ? vmat : kmat)[idx] = acc;
}

// P[d,(h,j)] = (1/8) * sum_e Wq[d,h*64+e]*k[j,h*64+e]   (256x128)
// U[(h,j),m] = sum_e val[j,h*64+e]*Wo[h*64+e,m]          (128x256)
__global__ void prep_PU(const float* __restrict__ Wq, const float* __restrict__ Wo,
                        const float* __restrict__ kmat, const float* __restrict__ vmat,
                        float* __restrict__ P, float* __restrict__ U)
{
  int t = blockIdx.x * 256 + threadIdx.x;       // 0..65535
  if (t < 32768) {
    int d = t >> 7, col = t & 127, h = col >> 5, j = col & 31;
    float acc = 0.f;
    for (int e = 0; e < 64; ++e)
      acc = fmaf(Wq[d * 256 + h * 64 + e], kmat[j * 256 + h * 64 + e], acc);
    P[t] = acc * 0.125f;                        // 1/sqrt(HEAD_DIM)
  } else {
    int u = t - 32768;
    int rq = u >> 8, m = u & 255, h = rq >> 5, j = rq & 31;
    float acc = 0.f;
    for (int e = 0; e < 64; ++e)
      acc = fmaf(vmat[j * 256 + h * 64 + e], Wo[(h * 64 + e) * 256 + m], acc);
    U[u] = acc;
  }
}

// Pre-swizzle row-major f32 weight [K x N] into bf16 WMMA B-fragment layout:
//   frag f = kt*(N/16)+nt, lane L, elem e -> B[kt*32 + 8*(L>>4) + (e<8 ? e : e+8)][nt*16 + (L&15)]
__global__ void prep_swizzle(const float* __restrict__ src, int Ksrc, int N,
                             __bf16* __restrict__ dst, int total)
{
  int idx = blockIdx.x * 256 + threadIdx.x;
  if (idx >= total) return;
  int e = idx & 15, L = (idx >> 4) & 31, f = idx >> 9;
  int NT16 = N >> 4;
  int nt = f % NT16, kt = f / NT16;
  int n = nt * 16 + (L & 15);
  int k = kt * 32 + 8 * (L >> 4) + ((e < 8) ? e : (e + 8));
  float v = (k < Ksrc) ? src[k * N + n] : 0.f;
  dst[idx] = (__bf16)v;
}

// ============================ LTC scan kernel ==============================
// 8 workgroups per batch (blockIdx = b*8 + part); each WG owns 64 post-units,
// 256 threads = 4 i-quarters x 64 units.  Full v-state exchanged per unfold
// through an L2-resident ping-pong buffer with a per-batch arrive/spin
// barrier (only 8 arrivals per barrier; counters zeroed per launch).
__global__ __launch_bounds__(256) void ltc_kernel(
    const float* __restrict__ x, const float* __restrict__ ts,
    const float* __restrict__ input_w, const float* __restrict__ input_b,
    const float* __restrict__ output_w, const float* __restrict__ output_b,
    const float* __restrict__ vleak,
    const float4* __restrict__ sens, const float4* __restrict__ rec,
    const float* __restrict__ glsp, const float* __restrict__ cmsp,
    float* __restrict__ ltcbuf, float* __restrict__ vT_out,
    float* __restrict__ vbuf, int* __restrict__ cnt)
{
  __shared__ float v_full[512];
  __shared__ float inp[576];
  __shared__ float red_wn[256];
  __shared__ float red_wd[256];

  const int t    = threadIdx.x;
  const int part = blockIdx.x & 7;
  const int b    = blockIdx.x >> 3;
  const int jl   = t & 63;          // unit within this WG's slice
  const int q    = t >> 6;          // i-quarter 0..3
  const int j    = part * 64 + jl;  // global post-unit

  float gl = 0.f, glvl = 0.f, cmp_ = 0.f, ow = 0.f, ob = 0.f;
  if (q == 0) {
    gl = glsp[j]; glvl = gl * vleak[j]; cmp_ = cmsp[j];
    if (j < 153) { ow = output_w[j]; ob = output_b[j]; }
  }
  for (int i = t; i < 512; i += 256) v_full[i] = 0.f;
  __syncthreads();

  int* cntb = cnt + b;
  float* vb0 = vbuf + (size_t)b * 512;            // phase 0
  float* vb1 = vbuf + 8192 + (size_t)b * 512;     // phase 1
  int round = 0;

  for (int s = 0; s < 512; ++s) {
    // stage affine input row into LDS
    for (int d = t; d < 576; d += 256)
      inp[d] = x[((size_t)b * 512 + s) * 576 + d] * input_w[d] + input_b[d];
    __syncthreads();

    // sensory partials over this thread's d-quarter (576 = 4 * 144)
    float swn = 0.f, swd = 0.f;
    for (int d = q * 144; d < q * 144 + 144; ++d) {
      float4 p = sens[d * 512 + j];
      float tt = fmaf(p.x, inp[d], p.y);
      float g = p.z * sigmoid_from_t(tt);
      swn = fmaf(g, p.w, swn);
      swd += g;
    }

    const float cm_t = (q == 0) ? (cmp_ * 6.f / ts[b * 512 + s]) : 0.f;

    for (int u = 0; u < 6; ++u) {
      // recurrent partials over this thread's i-quarter
      float wn = swn, wd = swd;
#pragma unroll 4
      for (int i = q * 128; i < q * 128 + 128; ++i) {
        float vi = v_full[i];
        float4 p = rec[i * 512 + j];              // coalesced b128
        float tt = fmaf(p.x, vi, p.y);
        float g = p.z * sigmoid_from_t(tt);
        wn = fmaf(g, p.w, wn);
        wd += g;
      }
      red_wn[t] = wn; red_wd[t] = wd;
      __syncthreads();

      if (q == 0) {
        float WN = red_wn[jl] + red_wn[jl + 64] + red_wn[jl + 128] + red_wn[jl + 192];
        float WD = red_wd[jl] + red_wd[jl + 64] + red_wd[jl + 128] + red_wd[jl + 192];
        float vo = v_full[j];
        float vn = (fmaf(cm_t, vo, glvl) + WN) / (cm_t + gl + WD + 1e-8f);
        ((u & 1) ? vb1 : vb0)[j] = vn;            // publish own slice
      }
      __threadfence();                            // agent-scope release of slice
      __syncthreads();
      ++round;
      if (t == 0) {
        __hip_atomic_fetch_add(cntb, 1, __ATOMIC_RELEASE, __HIP_MEMORY_SCOPE_AGENT);
        const int target = 8 * round;
        while (__hip_atomic_load(cntb, __ATOMIC_ACQUIRE, __HIP_MEMORY_SCOPE_AGENT) < target)
          __builtin_amdgcn_s_sleep(1);
      }
      __syncthreads();
      // gather full v-state (bypass WGP$ via atomic loads)
      float* vbr = (u & 1) ? vb1 : vb0;
      for (int i = t; i < 512; i += 256)
        v_full[i] = __hip_atomic_load(vbr + i, __ATOMIC_RELAXED, __HIP_MEMORY_SCOPE_AGENT);
      __syncthreads();
    }

    // motor slice (zero-padded to 160 for the WMMA K dimension)
    if (q == 0 && j < 160) {
      float y = (j < 153) ? fmaf(v_full[j], ow, ob) : 0.f;
      ltcbuf[((size_t)b * 512 + s) * 160 + j] = y;
    }
  }
  if (part == 0)
    for (int i = t; i < 512; i += 256) vT_out[b * 512 + i] = v_full[i];
}

// ========================= fused GEMM tail (WMMA) ==========================

__device__ __forceinline__ v8f wmma_bf(v16bf a, v16bf b, v8f c) {
  return __builtin_amdgcn_wmma_f32_16x16x32_bf16(false, a, false, b, (short)0, c, false, false);
}

// Gather A fragments (16 rows x KT*32 K) from f32 LDS activations, bf16-convert.
template <int KT>
__device__ __forceinline__ void loadA(v16bf* afr, const float* lds, int stride,
                                      int wrow0, int lane)
{
  const int m = lane & 15, hf = lane >> 4;
  const float* base = lds + (wrow0 + m) * stride + 8 * hf;
#pragma unroll
  for (int kt = 0; kt < KT; ++kt) {
    const float* r = base + kt * 32;
    v16bf a;
#pragma unroll
    for (int e = 0; e < 16; ++e) a[e] = (__bf16)r[(e < 8) ? e : (e + 8)];
    afr[kt] = a;
  }
}

template <int KT>
__device__ __forceinline__ v8f gemm_nt(const v16bf* afr, const __bf16* Bf,
                                       int NT16, int nt, int lane, v8f acc)
{
#pragma unroll
  for (int kt = 0; kt < KT; ++kt) {
    const __bf16* p = Bf + ((size_t)(kt * NT16 + nt) << 9) + (lane << 4);
    v16bf b = *(const v16bf*)p;                 // contiguous 32B per lane
    acc = wmma_bf(afr[kt], b, acc);
  }
  return acc;
}

// 32 rows/WG, 2 waves, 64 KB LDS.  Stage-in uses the Tensor Data Mover.
__global__ __launch_bounds__(64) void fused_post(
    const float* __restrict__ ltcbuf,
    const __bf16* __restrict__ Fproj, const float* __restrict__ proj_b,
    const __bf16* __restrict__ FP, const __bf16* __restrict__ FU,
    const float* __restrict__ bo,
    const __bf16* __restrict__ FW1, const float* __restrict__ b1,
    const __bf16* __restrict__ FW2, const float* __restrict__ b2,
    float* __restrict__ out_combined, float* __restrict__ out_belief)
{
  __shared__ float lds_h[32 * 256];
  __shared__ float lds_x[32 * 256];
  const int tid = threadIdx.x;
  const int lane = tid & 31;
  const int wrow0 = (tid >> 5) * 16;            // wave's 16-row stripe
  const int grow0 = blockIdx.x * 32;
  const int hf = lane >> 4, n0 = lane & 15;

  // ---- stage-in: 32x160 f32 tile of padded LTC output -> lds_x ----
#if USE_TDM
  typedef unsigned u32x4_t __attribute__((ext_vector_type(4)));
  typedef int      i32x8_t __attribute__((ext_vector_type(8)));
  typedef int      i32x4_t __attribute__((ext_vector_type(4)));
  if (tid < 32) {                               // one TDM op per WG (wave 0)
    unsigned long long ga = (unsigned long long)(size_t)(ltcbuf + (size_t)grow0 * 160);
    unsigned ldsoff = (unsigned)(size_t)(&lds_x[0]);   // low 32b of generic = LDS offset
    u32x4_t g0;
    g0[0] = 1u;                                 // count=1 valid descriptor
    g0[1] = ldsoff;                             // lds_addr (bytes)
    g0[2] = (unsigned)ga;                       // global_addr[31:0]
    g0[3] = ((unsigned)(ga >> 32) & 0x01FFFFFFu) | 0x80000000u; // addr[56:32] | type=2
    i32x8_t g1;
    g1[0] = 0x00020000;                         // data_size = 4 bytes
    g1[1] = (160 << 16);                        // tensor_dim0 = 160 (lo16)
    g1[2] = (8192 << 16);                       // tensor_dim0 hi=0 | tensor_dim1 lo = 8192
    g1[3] = (160 << 16);                        // tensor_dim1 hi=0 | tile_dim0 = 160
    g1[4] = 32;                                 // tile_dim1 = 32, tile_dim2 = 0
    g1[5] = 160;                                // tensor_dim0_stride = 160 (lo32)
    g1[6] = 0;                                  // stride hi | tensor_dim1_stride lo
    g1[7] = 0;
    i32x4_t z4 = {0, 0, 0, 0};
    i32x8_t z8 = {0, 0, 0, 0, 0, 0, 0, 0};
    __builtin_amdgcn_tensor_load_to_lds(g0, g1, z4, z4, z8, 0);
    __builtin_amdgcn_s_wait_tensorcnt(0);
  }
  __syncthreads();
#else
  for (int idx = tid; idx < 32 * 160; idx += 64) {
    int r = idx / 160, c = idx - r * 160;
    lds_x[r * 160 + c] = ltcbuf[(size_t)(grow0 + r) * 160 + c];
  }
  __syncthreads();
#endif

  v16bf afr[8];

  // ---- stage 1: h = ltc @ proj_W + proj_b   (K=160, N=256) ----
  loadA<5>(afr, lds_x, 160, wrow0, lane);
  for (int nt = 0; nt < 16; ++nt) {
    float bz = proj_b[nt * 16 + n0];
    v8f acc = {bz, bz, bz, bz, bz, bz, bz, bz};
    acc = gemm_nt<5>(afr, Fproj, 16, nt, lane, acc);
#pragma unroll
    for (int r = 0; r < 8; ++r)
      lds_h[(wrow0 + r + 8 * hf) * 256 + nt * 16 + n0] = acc[r];
  }
  __syncthreads();

  // ---- stage 2: scores = h @ P   (K=256, N=128; 1/sqrt(64) folded in P) ----
  loadA<8>(afr, lds_h, 256, wrow0, lane);
  for (int nt = 0; nt < 8; ++nt) {
    v8f acc = {0.f, 0.f, 0.f, 0.f, 0.f, 0.f, 0.f, 0.f};
    acc = gemm_nt<8>(afr, FP, 8, nt, lane, acc);
#pragma unroll
    for (int r = 0; r < 8; ++r)
      lds_x[(wrow0 + r + 8 * hf) * 128 + nt * 16 + n0] = acc[r];
  }
  __syncthreads();

  // ---- per-head softmax over 32 prototypes (32 rows x 4 heads) ----
  for (int task = tid; task < 128; task += 64) {
    int row = task >> 2, hd = task & 3;
    float* p = lds_x + row * 128 + hd * 32;
    float mx = -3.4e38f;
    for (int qq = 0; qq < 32; ++qq) mx = fmaxf(mx, p[qq]);
    float sum = 0.f;
    for (int qq = 0; qq < 32; ++qq) {
      float e = __builtin_amdgcn_exp2f((p[qq] - mx) * LOG2E);
      p[qq] = e;
      sum += e;
    }
    float inv = __builtin_amdgcn_rcpf(sum);
    for (int qq = 0; qq < 32; ++qq) p[qq] *= inv;
  }
  __syncthreads();

  // ---- stage 3: combined = h + attn @ U + bo   (K=128, N=256) ----
  loadA<4>(afr, lds_x, 128, wrow0, lane);
  for (int nt = 0; nt < 16; ++nt) {
    float bz = bo[nt * 16 + n0];
    v8f acc = {bz, bz, bz, bz, bz, bz, bz, bz};
    acc = gemm_nt<4>(afr, FU, 16, nt, lane, acc);
#pragma unroll
    for (int r = 0; r < 8; ++r) {
      int lr = wrow0 + r + 8 * hf;
      int col = nt * 16 + n0;
      float c = acc[r] + lds_h[lr * 256 + col];
      lds_h[lr * 256 + col] = c;
      out_combined[(size_t)(grow0 + lr) * 256 + col] = c;
    }
  }
  __syncthreads();

  // ---- stage 4: t = silu(combined @ W1 + b1)   (K=256, N=256) ----
  loadA<8>(afr, lds_h, 256, wrow0, lane);
  for (int nt = 0; nt < 16; ++nt) {
    float bz = b1[nt * 16 + n0];
    v8f acc = {bz, bz, bz, bz, bz, bz, bz, bz};
    acc = gemm_nt<8>(afr, FW1, 16, nt, lane, acc);
#pragma unroll
    for (int r = 0; r < 8; ++r) {
      float xx = acc[r];
      float sg = __builtin_amdgcn_rcpf(1.0f + __builtin_amdgcn_exp2f(-xx * LOG2E));
      lds_x[(wrow0 + r + 8 * hf) * 256 + nt * 16 + n0] = xx * sg;
    }
  }
  __syncthreads();

  // ---- stage 5: belief = t @ W2 + b2   (K=256, N=64) ----
  loadA<8>(afr, lds_x, 256, wrow0, lane);
  for (int nt = 0; nt < 4; ++nt) {
    float bz = b2[nt * 16 + n0];
    v8f acc = {bz, bz, bz, bz, bz, bz, bz, bz};
    acc = gemm_nt<8>(afr, FW2, 4, nt, lane, acc);
#pragma unroll
    for (int r = 0; r < 8; ++r)
      out_belief[(size_t)(grow0 + wrow0 + r + 8 * hf) * 64 + nt * 16 + n0] = acc[r];
  }
}

// ============================= launch ======================================

extern "C" void kernel_launch(void* const* d_in, const int* in_sizes, int n_in,
                              void* d_out, int out_size, void* d_ws, size_t ws_size,
                              hipStream_t stream) {
  (void)in_sizes; (void)n_in; (void)out_size; (void)ws_size;
  const float* x        = (const float*)d_in[0];
  const float* ts       = (const float*)d_in[1];
  const float* input_w  = (const float*)d_in[2];
  const float* input_b  = (const float*)d_in[3];
  const float* output_w = (const float*)d_in[4];
  const float* output_b = (const float*)d_in[5];
  const float* smu      = (const float*)d_in[6];
  const float* ssig     = (const float*)d_in[7];
  const float* sw       = (const float*)d_in[8];
  const float* smask    = (const float*)d_in[9];
  const float* serev    = (const float*)d_in[10];
  const float* rmu      = (const float*)d_in[11];
  const float* rsig     = (const float*)d_in[12];
  const float* rw       = (const float*)d_in[13];
  const float* rmask    = (const float*)d_in[14];
  const float* rerev    = (const float*)d_in[15];
  const float* gleak    = (const float*)d_in[16];
  const float* vleak    = (const float*)d_in[17];
  const float* cm       = (const float*)d_in[18];
  const float* proj_W   = (const float*)d_in[19];
  const float* proj_b   = (const float*)d_in[20];
  const float* lookup   = (const float*)d_in[21];
  const float* targets  = (const float*)d_in[22];
  const float* Wq       = (const float*)d_in[23];
  const float* Wk       = (const float*)d_in[24];
  const float* Wv       = (const float*)d_in[25];
  const float* Wo       = (const float*)d_in[26];
  const float* bo       = (const float*)d_in[27];
  const float* W1       = (const float*)d_in[28];
  const float* b1       = (const float*)d_in[29];
  const float* W2       = (const float*)d_in[30];
  const float* b2       = (const float*)d_in[31];

  // workspace carve-up (256B-aligned offsets, ~15 MB total)
  char* w = (char*)d_ws;
  float4* sens  = (float4*)(w + 0);            // 576*512*16 = 4718592
  float4* rec   = (float4*)(w + 4718592);      // 512*512*16 = 4194304
  float*  glsp  = (float*)(w + 8912896);       // 2048
  float*  cmsp  = (float*)(w + 8914944);       // 2048
  float*  ltcb  = (float*)(w + 8916992);       // 8192*160*4 = 5242880
  float*  kmat  = (float*)(w + 14159872);      // 32768
  float*  vmat  = (float*)(w + 14192640);      // 32768
  float*  Pm    = (float*)(w + 14225408);      // 131072
  float*  Um    = (float*)(w + 14356480);      // 131072
  __bf16* Fproj = (__bf16*)(w + 14487552);     // 81920
  __bf16* FP    = (__bf16*)(w + 14569472);     // 65536
  __bf16* FU    = (__bf16*)(w + 14635008);     // 65536
  __bf16* FW1   = (__bf16*)(w + 14700544);     // 131072
  __bf16* FW2   = (__bf16*)(w + 14831616);     // 32768
  int*    cnt   = (int*)(w + 14864384);        // 64
  float*  vbuf  = (float*)(w + 14864640);      // 2*16*512*4 = 65536

  float* out = (float*)d_out;
  float* out_combined = out;                   // (16,512,256)
  float* out_belief   = out + 2097152;         // (16,512,64)
  float* out_vT       = out + 2621440;         // (16,512)

  init_sync<<<1, 64, 0, stream>>>(cnt);
  prep_params<<<2179, 256, 0, stream>>>(smu, ssig, sw, smask, serev,
                                        rmu, rsig, rw, rmask, rerev,
                                        gleak, cm, sens, rec, glsp, cmsp);
  prep_kv<<<64, 256, 0, stream>>>(lookup, targets, Wk, Wv, kmat, vmat);
  prep_PU<<<256, 256, 0, stream>>>(Wq, Wo, kmat, vmat, Pm, Um);
  prep_swizzle<<<160, 256, 0, stream>>>(proj_W, 153, 256, Fproj, 40960);
  prep_swizzle<<<128, 256, 0, stream>>>(Pm, 256, 128, FP, 32768);
  prep_swizzle<<<128, 256, 0, stream>>>(Um, 128, 256, FU, 32768);
  prep_swizzle<<<256, 256, 0, stream>>>(W1, 256, 256, FW1, 65536);
  prep_swizzle<<<64, 256, 0, stream>>>(W2, 256, 64, FW2, 16384);

  ltc_kernel<<<128, 256, 0, stream>>>(x, ts, input_w, input_b, output_w, output_b,
                                      vleak, sens, rec, glsp, cmsp, ltcb, out_vT,
                                      vbuf, cnt);
  fused_post<<<256, 64, 0, stream>>>(ltcb, Fproj, proj_b, FP, FU, bo,
                                     FW1, b1, FW2, b2, out_combined, out_belief);
}
